// MultiHeadAttention_58548994179294
// MI455X (gfx1250) — matrix operand order repaired and checked by reading, not compiled, not run on hardware
//
#include <hip/hip_runtime.h>

#define HN    16
#define DIN   1024
#define DH    64
#define BATCH 2
#define SEQ   2048
#define ROWS  (BATCH * SEQ)   // 4096

typedef __bf16 bf16;
typedef __attribute__((ext_vector_type(16))) bf16  bf16x16;
typedef __attribute__((ext_vector_type(8)))  bf16  bf16x8;
typedef __attribute__((ext_vector_type(8)))  float v8f;
typedef int v4i __attribute__((vector_size(16)));

union ABFrag { bf16x16 v; bf16x8 h[2]; };

#if __has_builtin(__builtin_amdgcn_global_load_async_to_lds_b128)
#define HAVE_ASYNC_LDS 1
#else
#define HAVE_ASYNC_LDS 0
#endif

__device__ __forceinline__ v8f wmma_bf16(bf16x16 a, bf16x16 b, v8f c) {
  return __builtin_amdgcn_wmma_f32_16x16x32_bf16(false, a, false, b, (short)0, c, false, false);
}

__device__ __forceinline__ float redmax16(float v) {
  v = fmaxf(v, __shfl_xor(v, 1, 16));
  v = fmaxf(v, __shfl_xor(v, 2, 16));
  v = fmaxf(v, __shfl_xor(v, 4, 16));
  v = fmaxf(v, __shfl_xor(v, 8, 16));
  return v;
}
__device__ __forceinline__ float redsum16(float v) {
  v += __shfl_xor(v, 1, 16);
  v += __shfl_xor(v, 2, 16);
  v += __shfl_xor(v, 4, 16);
  v += __shfl_xor(v, 8, 16);
  return v;
}

// 16-byte global -> LDS copy (async on CDNA5, register fallback otherwise)
__device__ __forceinline__ void copy16_g2l(const bf16* gsrc, bf16* ldst) {
#if HAVE_ASYNC_LDS
  __builtin_amdgcn_global_load_async_to_lds_b128(
      (__attribute__((address_space(1))) v4i*)gsrc,
      (__attribute__((address_space(3))) v4i*)ldst, 0, 0);
#else
  *(bf16x8*)ldst = *(const bf16x8*)gsrc;
#endif
}

__device__ __forceinline__ void wait_async_le2() {
#if HAVE_ASYNC_LDS
#if __has_builtin(__builtin_amdgcn_s_wait_asynccnt)
  __builtin_amdgcn_s_wait_asynccnt(2);
#else
  asm volatile("s_wait_asynccnt 0x2" ::: "memory");
#endif
#else
  asm volatile("s_wait_dscnt 0x0" ::: "memory");
#endif
}

// ---------------- conversion / layout kernels ----------------

__global__ void cvt_bf16_kernel(const float* __restrict__ in, bf16* __restrict__ out, int n) {
  int i = blockIdx.x * blockDim.x + threadIdx.x;
  if (i < n) out[i] = (bf16)in[i];
}

// in [HN][DIN][DH] fp32 -> out [HN][DH][DIN] bf16
__global__ void transpose_w_kernel(const float* __restrict__ in, bf16* __restrict__ out, int n) {
  int i = blockIdx.x * blockDim.x + threadIdx.x;
  if (i >= n) return;
  int d = i % DH;
  int t = i / DH;
  int k = t % DIN;
  int h = t / DIN;
  out[((size_t)h * DH + d) * DIN + k] = (bf16)in[i];
}

// in [HN*DH][DIN] fp32 -> out [DIN][HN*DH] bf16
__global__ void transpose_wo_kernel(const float* __restrict__ in, bf16* __restrict__ out, int n) {
  int i = blockIdx.x * blockDim.x + threadIdx.x;
  if (i >= n) return;
  int c = i % DIN;
  int e = i / DIN;
  out[(size_t)c * (HN * DH) + e] = (bf16)in[i];
}

// ---------------- QKV projection: per-wave 16x64 tile, A-fragment reused across 4 WMMAs ----------------

__global__ __launch_bounds__(256) void qkv_proj_kernel(
    const bf16* __restrict__ xq, const bf16* __restrict__ xk, const bf16* __restrict__ xv,
    const bf16* __restrict__ wqT, const bf16* __restrict__ wkT, const bf16* __restrict__ wvT,
    const float* __restrict__ bq, const float* __restrict__ bk_, const float* __restrict__ bv,
    bf16* __restrict__ qb, bf16* __restrict__ kb, bf16* __restrict__ vt)
{
  int wave = threadIdx.x >> 5;
  int lane = threadIdx.x & 31;
  int tile = blockIdx.x * 8 + wave;
  int rt = tile & 255; tile >>= 8;   // 256 row tiles of 16 over ROWS
  int h  = tile & 15;
  int p  = tile >> 4;                // 0=q 1=k 2=v

  const bf16*  X    = (p == 0) ? xq : (p == 1) ? xk : xv;
  const bf16*  WT   = ((p == 0) ? wqT : (p == 1) ? wkT : wvT) + (size_t)h * DH * DIN;
  const float* bias = ((p == 0) ? bq : (p == 1) ? bk_ : bv) + h * DH;

  int ln = lane & 15, hi = lane >> 4, hi8 = hi * 8, hi16 = hi * 16;
  const bf16* xrow = X + (size_t)(rt * 16 + ln) * DIN;

  v8f c[4] = {{}, {}, {}, {}};
  for (int k0 = 0; k0 < DIN; k0 += 32) {
    ABFrag a;
    a.h[0] = *(const bf16x8*)(xrow + k0 + hi8);
    a.h[1] = *(const bf16x8*)(xrow + k0 + 16 + hi8);
#pragma unroll
    for (int ct = 0; ct < 4; ++ct) {
      const bf16* wrow = WT + (size_t)(ct * 16 + ln) * DIN + k0 + hi8;
      ABFrag bm;
      bm.h[0] = *(const bf16x8*)(wrow);
      bm.h[1] = *(const bf16x8*)(wrow + 16);
      c[ct] = wmma_bf16(a.v, bm.v, c[ct]);
    }
  }

#pragma unroll
  for (int ct = 0; ct < 4; ++ct) {
    float bval = bias[ct * 16 + ln];
#pragma unroll
    for (int r = 0; r < 8; ++r) c[ct][r] += bval;
  }

  if (p == 2) {
    // store V transposed: vt[(h*64+d)*ROWS + row]; 8 consecutive rows per lane -> one b128 store
#pragma unroll
    for (int ct = 0; ct < 4; ++ct) {
      union { bf16x8 v; bf16 e[8]; } pk;
#pragma unroll
      for (int r = 0; r < 8; ++r) pk.e[r] = (bf16)c[ct][r];
      bf16* vp = vt + (size_t)(h * DH + ct * 16 + ln) * ROWS + rt * 16 + hi8;
      *(bf16x8*)vp = pk.v;
    }
  } else {
    bf16* O = ((p == 0) ? qb : kb) + (size_t)h * ROWS * DH;
#pragma unroll
    for (int ct = 0; ct < 4; ++ct)
#pragma unroll
      for (int r = 0; r < 8; ++r)
        O[(size_t)(rt * 16 + r + hi8) * DH + ct * 16 + ln] = (bf16)c[ct][r];
  }
}

// ---------------- flash attention: block shares K/V tiles in LDS (async double-buffered) ----------------

__global__ __launch_bounds__(256) void attn_kernel(
    const bf16* __restrict__ qb, const bf16* __restrict__ kb,
    const bf16* __restrict__ vt, bf16* __restrict__ catb)
{
  __shared__ bf16 kbuf[2][32 * DH];   // 32 keys x 64 dh      (2 x 4KB)
  __shared__ bf16 vbuf[2][DH * 32];   // 64 dh  x 32 keys     (2 x 4KB)
  __shared__ bf16 ptile[8][16 * 32];  // per-wave P tile      (8KB)

  int tid  = threadIdx.x;
  int wave = tid >> 5;
  int lane = tid & 31;
  int blk  = blockIdx.x;
  int qg = blk & 15;            // 16 query groups of 128 rows
  int b  = (blk >> 4) & 1;
  int h  = blk >> 5;
  int qt = qg * 8 + wave;       // this wave's 16-row query tile
  int ln = lane & 15, hi = lane >> 4, hi8 = hi * 8, hi16 = hi * 16;

  size_t qkbase = ((size_t)h * ROWS + (size_t)b * SEQ) * DH;
  const bf16* kbB = kb + qkbase;
  const bf16* vtB = vt + (size_t)h * DH * ROWS + (size_t)b * SEQ;

  // cooperative staging indices: one b128 per thread per tile
  int kr = tid >> 3, kc = (tid & 7) * 8;   // K: 32 rows x (8 chunks of 8 bf16)
  int vr = tid >> 2, vc = (tid & 3) * 8;   // V: 64 rows x (4 chunks of 8 bf16)

  // Q A-fragments for K-dim 0..31 and 32..63 (held in registers for entire pass)
  ABFrag aq0, aq1;
  {
    const bf16* qp = qb + qkbase + (size_t)(qt * 16 + ln) * DH;
    aq0.h[0] = *(const bf16x8*)(qp + hi8);
    aq0.h[1] = *(const bf16x8*)(qp + 16 + hi8);
    aq1.h[0] = *(const bf16x8*)(qp + 32 + hi8);
    aq1.h[1] = *(const bf16x8*)(qp + 48 + hi8);
  }

  v8f o[4] = {{}, {}, {}, {}};
  float mrow[8], lrow[8];
#pragma unroll
  for (int r = 0; r < 8; ++r) { mrow[r] = -3.0e38f; lrow[r] = 0.0f; }

  const int NSTEPS = SEQ / 32;

  // prologue: issue stages 0 and 1
#pragma unroll
  for (int pre = 0; pre < 2; ++pre) {
    int j = pre * 32;
    copy16_g2l(kbB + (size_t)(j + kr) * DH + kc, &kbuf[pre][kr * DH + kc]);
    copy16_g2l(vtB + (size_t)vr * ROWS + j + vc, &vbuf[pre][vr * 32 + vc]);
  }

  for (int jt = 0; jt < NSTEPS; ++jt) {
    int buf = jt & 1;
    wait_async_le2();             // stage jt complete (stage jt+1 may remain in flight)
    __syncthreads();

    const bf16* kT = kbuf[buf];
    const bf16* vT = vbuf[buf];

    // scores: q(16x64) @ k^T(64x32) -> two 16x16 tiles, K/V fragments from LDS
    v8f s0 = {}, s1 = {};
    {
      ABFrag bkf;
      bkf.h[0] = *(const bf16x8*)(kT + ln * DH + hi8);
      bkf.h[1] = *(const bf16x8*)(kT + ln * DH + 16 + hi8);
      s0 = wmma_bf16(aq0.v, bkf.v, s0);
      bkf.h[0] = *(const bf16x8*)(kT + ln * DH + 32 + hi8);
      bkf.h[1] = *(const bf16x8*)(kT + ln * DH + 48 + hi8);
      s0 = wmma_bf16(aq1.v, bkf.v, s0);
      bkf.h[0] = *(const bf16x8*)(kT + (16 + ln) * DH + hi8);
      bkf.h[1] = *(const bf16x8*)(kT + (16 + ln) * DH + 16 + hi8);
      s1 = wmma_bf16(aq0.v, bkf.v, s1);
      bkf.h[0] = *(const bf16x8*)(kT + (16 + ln) * DH + 32 + hi8);
      bkf.h[1] = *(const bf16x8*)(kT + (16 + ln) * DH + 48 + hi8);
      s1 = wmma_bf16(aq1.v, bkf.v, s1);
    }

    // online softmax; C-fragment rows are r+8*hi, columns spread over 16 lanes of each half
    bf16* pw = ptile[wave];
#pragma unroll
    for (int r = 0; r < 8; ++r) {
      float a  = s0[r] * 0.125f;      // 1/sqrt(64)
      float c2 = s1[r] * 0.125f;
      float sm   = redmax16(fmaxf(a, c2));
      float newm = fmaxf(mrow[r], sm);
      float p0   = __expf(a - newm);
      float p1   = __expf(c2 - newm);
      float corr = __expf(mrow[r] - newm);
      float rs   = redsum16(p0 + p1);
      lrow[r] = lrow[r] * corr + rs;
      mrow[r] = newm;
      o[0][r] *= corr; o[1][r] *= corr; o[2][r] *= corr; o[3][r] *= corr;
      pw[(r + hi8) * 32 + ln]      = (bf16)p0;   // C layout -> row-major 16x32
      pw[(r + hi8) * 32 + 16 + ln] = (bf16)p1;
    }
    asm volatile("s_wait_dscnt 0x0" ::: "memory");

    // reload P as A-fragment (16x32)
    ABFrag ap;
    ap.h[0] = *(const bf16x8*)(pw + ln * 32 + hi8);
    ap.h[1] = *(const bf16x8*)(pw + ln * 32 + 16 + hi8);

    // P(16x32) @ V(32x64): 4 dv tiles from shared transposed-V tile
#pragma unroll
    for (int d4 = 0; d4 < 4; ++d4) {
      const bf16* vp = vT + (d4 * 16 + ln) * 32 + hi8;
      ABFrag bvf;
      bvf.h[0] = *(const bf16x8*)(vp);
      bvf.h[1] = *(const bf16x8*)(vp + 16);
      o[d4] = wmma_bf16(ap.v, bvf.v, o[d4]);
    }

    __syncthreads();              // everyone done reading buf before it is overwritten

    int jn = (jt + 2) * 32;
    if (jn < SEQ) {               // issue stage jt+2 into this buffer
      copy16_g2l(kbB + (size_t)(jn + kr) * DH + kc, &kbuf[buf][kr * DH + kc]);
      copy16_g2l(vtB + (size_t)vr * ROWS + jn + vc, &vbuf[buf][vr * 32 + vc]);
    }
  }

  // normalize and scatter into cat[b][s][h*64+d] (bf16)
#pragma unroll
  for (int r = 0; r < 8; ++r) {
    float inv = 1.0f / lrow[r];
    int srow = qt * 16 + r + hi8;
    bf16* cp = catb + ((size_t)b * SEQ + srow) * (HN * DH) + h * DH + ln;
    cp[0]  = (bf16)(o[0][r] * inv);
    cp[16] = (bf16)(o[1][r] * inv);
    cp[32] = (bf16)(o[2][r] * inv);
    cp[48] = (bf16)(o[3][r] * inv);
  }
}

// ---------------- output projection: per-wave 16x64 tile of cat[4096x1024] @ Wo + bo ----------------

__global__ __launch_bounds__(256) void oproj_kernel(
    const bf16* __restrict__ catb, const bf16* __restrict__ woT,
    const float* __restrict__ bo, float* __restrict__ out)
{
  int wave = threadIdx.x >> 5;
  int lane = threadIdx.x & 31;
  int tile = blockIdx.x * 8 + wave;
  int cg = tile & 15;    // 16 column groups of 64 over DIN
  int rt = tile >> 4;    // 256 row tiles of 16 over ROWS
  int ln = lane & 15, hi = lane >> 4, hi8 = hi * 8, hi16 = hi * 16;

  const bf16* arow = catb + (size_t)(rt * 16 + ln) * (HN * DH);

  v8f c[4] = {{}, {}, {}, {}};
  for (int k0 = 0; k0 < HN * DH; k0 += 32) {
    ABFrag a;
    a.h[0] = *(const bf16x8*)(arow + k0 + hi8);
    a.h[1] = *(const bf16x8*)(arow + k0 + 16 + hi8);
#pragma unroll
    for (int ct = 0; ct < 4; ++ct) {
      const bf16* brow = woT + (size_t)((cg * 4 + ct) * 16 + ln) * (HN * DH) + k0 + hi8;
      ABFrag bm;
      bm.h[0] = *(const bf16x8*)(brow);
      bm.h[1] = *(const bf16x8*)(brow + 16);
      c[ct] = wmma_bf16(a.v, bm.v, c[ct]);
    }
  }

#pragma unroll
  for (int ct = 0; ct < 4; ++ct) {
    int col = (cg * 4 + ct) * 16 + ln;
    float bval = bo[col];
#pragma unroll
    for (int r = 0; r < 8; ++r)
      out[(size_t)(rt * 16 + r + hi8) * DIN + col] = c[ct][r] + bval;
  }
}

// ---------------- host launch ----------------

extern "C" void kernel_launch(void* const* d_in, const int* in_sizes, int n_in,
                              void* d_out, int out_size, void* d_ws, size_t ws_size,
                              hipStream_t stream) {
  (void)in_sizes; (void)n_in; (void)out_size; (void)ws_size;

  const float* query = (const float*)d_in[0];
  const float* key_  = (const float*)d_in[1];
  const float* value = (const float*)d_in[2];
  const float* Wq    = (const float*)d_in[3];
  const float* bq    = (const float*)d_in[4];
  const float* Wk    = (const float*)d_in[5];
  const float* bk    = (const float*)d_in[6];
  const float* Wv    = (const float*)d_in[7];
  const float* bv    = (const float*)d_in[8];
  const float* Wo    = (const float*)d_in[9];
  const float* bo    = (const float*)d_in[10];

  char* w = (char*)d_ws;
  auto alloc = [&](size_t bytes) -> void* {
    void* p = (void*)w;
    w += (bytes + 255) & ~(size_t)255;
    return p;
  };

  bf16* xq   = (bf16*)alloc((size_t)ROWS * DIN * 2);
  bf16* xk   = (bf16*)alloc((size_t)ROWS * DIN * 2);
  bf16* xv   = (bf16*)alloc((size_t)ROWS * DIN * 2);
  bf16* wqT  = (bf16*)alloc((size_t)HN * DH * DIN * 2);
  bf16* wkT  = (bf16*)alloc((size_t)HN * DH * DIN * 2);
  bf16* wvT  = (bf16*)alloc((size_t)HN * DH * DIN * 2);
  bf16* woT  = (bf16*)alloc((size_t)DIN * HN * DH * 2);
  bf16* qb   = (bf16*)alloc((size_t)HN * ROWS * DH * 2);
  bf16* kbuf = (bf16*)alloc((size_t)HN * ROWS * DH * 2);
  bf16* vt   = (bf16*)alloc((size_t)HN * DH * ROWS * 2);
  bf16* catb = (bf16*)alloc((size_t)ROWS * HN * DH * 2);

  const int n_x = ROWS * DIN;                  // 4,194,304
  cvt_bf16_kernel<<<(n_x + 255) / 256, 256, 0, stream>>>(query, xq, n_x);
  cvt_bf16_kernel<<<(n_x + 255) / 256, 256, 0, stream>>>(key_,  xk, n_x);
  cvt_bf16_kernel<<<(n_x + 255) / 256, 256, 0, stream>>>(value, xv, n_x);

  const int n_w = HN * DIN * DH;               // 1,048,576
  transpose_w_kernel<<<(n_w + 255) / 256, 256, 0, stream>>>(Wq, wqT, n_w);
  transpose_w_kernel<<<(n_w + 255) / 256, 256, 0, stream>>>(Wk, wkT, n_w);
  transpose_w_kernel<<<(n_w + 255) / 256, 256, 0, stream>>>(Wv, wvT, n_w);
  const int n_wo = HN * DH * DIN;
  transpose_wo_kernel<<<(n_wo + 255) / 256, 256, 0, stream>>>(Wo, woT, n_wo);

  // 3 proj * 16 heads * 256 row-tiles = 12288 wave-tiles, 8 waves/block
  qkv_proj_kernel<<<(3 * HN * 256) / 8, 256, 0, stream>>>(
      xq, xk, xv, wqT, wkT, wvT, bq, bk, bv, qb, kbuf, vt);

  // 16 heads * 2 batch * 16 query-groups = 512 blocks (8 query tiles each)
  attn_kernel<<<HN * BATCH * 16, 256, 0, stream>>>(qb, kbuf, vt, catb);

  // 256 row-tiles * 16 col-groups = 4096 wave-tiles
  oproj_kernel<<<(256 * 16) / 8, 256, 0, stream>>>(catb, woT, bo, (float*)d_out);
}
